// Self_paced_loss_21474836480579
// MI455X (gfx1250) — matrix operand order repaired and checked by reading, not compiled, not run on hardware
//
#include <hip/hip_runtime.h>

// ---------------------------------------------------------------------------
// loss = sum_ij S_ij * (||f_i||^2 + ||f_j||^2 - 2 f_i.f_j)
//
// MI455X strategy: single streaming pass over S (256 MB -> ~11us @ 23.3TB/s),
// fused bf16 WMMA GEMM  C = S @ [F | 1 | sq | 0pad]  (8192 x 144), and
// immediate per-wave linear collapse of C into a scalar (loss is linear in C,
// so K-slicing needs no materialized C and no atomics). S is read at full
// fp32 precision (the bandwidth floor anyway) and converted to bf16
// in-register via native v_cvt; S loads are non-temporal so the one-shot
// stream does not evict the L2-resident swizzled B operand.
// ---------------------------------------------------------------------------

typedef __attribute__((ext_vector_type(16))) __bf16          v16bf;
typedef __attribute__((ext_vector_type(8)))  float           v8f;
typedef __attribute__((ext_vector_type(4)))  float           v4f;

#define NN            8192
#define DD            128
#define CT            9                 // 144 cols = 9 tiles of 16
#define KTILE         32                // bf16 WMMA K
#define NJT           (NN / KTILE)      // 256 K-tiles
#define KSLICES       8
#define JT_PER_SLICE  (NJT / KSLICES)   // 32
#define ROWS_PER_WAVE 16
#define WAVES_PER_WG  8
#define ROWS_PER_WG   (ROWS_PER_WAVE * WAVES_PER_WG)  // 128
#define NIB           (NN / ROWS_PER_WG)              // 64 row blocks
#define NPART         (KSLICES * NIB * WAVES_PER_WG)  // 4096 partials

// --- kernel 1: sq[i] = ||F[i,:]||^2 ---------------------------------------
__global__ void k_sq(const float* __restrict__ F, float* __restrict__ sq) {
    int i = blockIdx.x * blockDim.x + threadIdx.x;      // NN threads
    const v4f* fr = (const v4f*)(F + (size_t)i * DD);
    float s = 0.f;
#pragma unroll
    for (int q = 0; q < DD / 4; ++q) {
        v4f v = fr[q];
        s += v[0]*v[0] + v[1]*v[1] + v[2]*v[2] + v[3]*v[3];
    }
    sq[i] = s;
}

// --- kernel 2: build B in WMMA-lane-swizzled bf16 layout -------------------
// Bsw tile (jt,ct), lane ln holds 16 bf16: col n = ct*16 + (ln&15),
// k = (ln>>4)*8 + p (p<8), 16 + (ln>>4)*8 + (p-8) (p>=8), j = jt*32 + k.   value = F[j,n] | 1 | sq[j] | 0.
__global__ void k_bsw(const float* __restrict__ F, const float* __restrict__ sq,
                      __bf16* __restrict__ Bsw) {
    int tile = blockIdx.x;              // 0 .. NJT*CT-1
    int jt = tile / CT, ct = tile % CT;
    int ln = threadIdx.x;               // 0..31
    int n  = ct * 16 + (ln & 15);
    int kh = ln >> 4;
    v16bf val;
#pragma unroll
    for (int p = 0; p < 16; ++p) {
        int j = jt * KTILE + (p < 8 ? kh * 8 + p : 16 + kh * 8 + (p - 8));
        float v;
        if      (n <  DD)     v = F[(size_t)j * DD + n];
        else if (n == DD)     v = 1.0f;
        else if (n == DD + 1) v = sq[j];
        else                  v = 0.0f;
        val[p] = (__bf16)v;
    }
    *(v16bf*)(Bsw + ((size_t)tile * 32 + ln) * 16) = val;
}

// --- kernel 3: streaming S-pass GEMM + fused linear collapse ---------------
__global__ void __launch_bounds__(256)
k_gemm_loss(const float* __restrict__ S, const float* __restrict__ F,
            const float* __restrict__ sq, const __bf16* __restrict__ Bsw,
            float* __restrict__ partials) {
    int ib   = blockIdx.x;              // row block  0..NIB-1
    int ks   = blockIdx.y;              // K slice    0..KSLICES-1
    int wave = threadIdx.x >> 5;        // wave32
    int ln   = threadIdx.x & 31;
    int ibase = ib * ROWS_PER_WG + wave * ROWS_PER_WAVE;

    // A-matrix lane mapping (16-bit A 16x32): row = ibase + (ln&15),
    // lane<16 -> K in {0..7,16..23}; lane>=16 -> K in {8..15,24..31}
    int arow = ibase + (ln & 15);
    int koff = (ln >> 4) * 8;
    const float* srow = S + (size_t)arow * NN;

    v8f acc[CT] = {};

    int jt0 = ks * JT_PER_SLICE;
#pragma unroll 1
    for (int jt = jt0; jt < jt0 + JT_PER_SLICE; ++jt) {
        int jbase = jt * KTILE;
        // keep the one-shot S stream ahead of the math (global_prefetch_b8)
        __builtin_prefetch(srow + jbase + 4 * KTILE, 0, 0);

        // non-temporal: S bytes are used exactly once
        v4f a0 = __builtin_nontemporal_load((const v4f*)(srow + jbase + koff));
        v4f a1 = __builtin_nontemporal_load((const v4f*)(srow + jbase + koff + 4));
        v4f a2 = __builtin_nontemporal_load((const v4f*)(srow + jbase + 16 + koff));
        v4f a3 = __builtin_nontemporal_load((const v4f*)(srow + jbase + 16 + koff + 4));

        v16bf A;
#pragma unroll
        for (int p = 0; p < 4; ++p) {   // native cvt -> v_cvt_pk_bf16_f32
            A[p]      = (__bf16)a0[p];
            A[4 + p]  = (__bf16)a1[p];
            A[8 + p]  = (__bf16)a2[p];
            A[12 + p] = (__bf16)a3[p];
        }

        const v16bf* bptr = (const v16bf*)Bsw + ((size_t)jt * CT * 32) + ln;
#pragma unroll
        for (int t = 0; t < CT; ++t) {
            v16bf B = bptr[(size_t)t * 32];
            // A is shared by all 9 WMMAs: set reuse_a on every instruction
            // that follows an identical one (ISA 7.12 OPSEL RA rule).
            if (t == 0)
                acc[t] = __builtin_amdgcn_wmma_f32_16x16x32_bf16(
                    false, A, false, B, (short)0, acc[t], false, false);
            else
                acc[t] = __builtin_amdgcn_wmma_f32_16x16x32_bf16(
                    false, A, false, B, (short)0, acc[t], true, false);
        }
    }

    // Epilogue: partial loss = sum of this wave's C entries * weights.
    // C layout: VGPR r on lane ln -> C[ibase + 8*(ln>>4) + r, 16*t + (ln&15)]
    int nloc = ln & 15;
    int kh   = ln >> 4;
    float s = 0.f;
#pragma unroll
    for (int r = 0; r < 8; ++r) {
        int i = ibase + kh * 8 + r;
        const float* frow = F + (size_t)i * DD;
#pragma unroll
        for (int t = 0; t < 8; ++t)
            s += acc[t][r] * (-2.0f) * frow[t * 16 + nloc];
        float w8 = (nloc == 0) ? sq[i] : ((nloc == 1) ? 1.0f : 0.0f);
        s += acc[8][r] * w8;
    }
    // fixed-order wave reduction (deterministic)
#pragma unroll
    for (int off = 16; off > 0; off >>= 1)
        s += __shfl_down(s, off, 32);
    if (ln == 0)
        partials[((size_t)ks * NIB + ib) * WAVES_PER_WG + wave] = s;
}

// --- kernel 4: deterministic final reduction -------------------------------
__global__ void k_reduce(const float* __restrict__ partials, float* __restrict__ out) {
    __shared__ float sm[256];
    float s = 0.f;
    for (int idx = threadIdx.x; idx < NPART; idx += 256)
        s += partials[idx];
    sm[threadIdx.x] = s;
    __syncthreads();
    for (int str = 128; str > 0; str >>= 1) {
        if ((int)threadIdx.x < str) sm[threadIdx.x] += sm[threadIdx.x + str];
        __syncthreads();
    }
    if (threadIdx.x == 0) out[0] = sm[0];
}

// ---------------------------------------------------------------------------
extern "C" void kernel_launch(void* const* d_in, const int* in_sizes, int n_in,
                              void* d_out, int out_size, void* d_ws, size_t ws_size,
                              hipStream_t stream) {
    (void)in_sizes; (void)n_in; (void)out_size; (void)ws_size;
    const float* F = (const float*)d_in[0];
    const float* S = (const float*)d_in[1];
    // d_in[2]=t, d_in[3]=T: loss does not depend on them.

    // workspace layout: [ sq: 8192 f32 | partials: 4096 f32 | pad | Bsw bf16 ]
    float*  sq       = (float*)d_ws;
    float*  partials = sq + NN;
    __bf16* Bsw      = (__bf16*)((char*)d_ws + 65536);
    float*  out      = (float*)d_out;

    k_sq   <<<NN / 256, 256, 0, stream>>>(F, sq);
    k_bsw  <<<NJT * CT, 32, 0, stream>>>(F, sq, Bsw);
    dim3 grid(NIB, KSLICES);
    k_gemm_loss<<<grid, 256, 0, stream>>>(S, F, sq, Bsw, partials);
    k_reduce<<<1, 256, 0, stream>>>(partials, out);
}
